// DifferentialFlashAttention_45595372814681
// MI455X (gfx1250) — matrix operand order repaired and checked by reading, not compiled, not run on hardware
//
#include <hip/hip_runtime.h>
#include <hip/hip_bf16.h>
#include <math.h>

// ---------------------------------------------------------------------------
// Differential Flash Attention for MI455X (gfx1250, wave32, WMMA bf16).
//
// Pipeline:
//   1) lam_kernel        : lam_full = exp(sum lq1*lk1) - exp(sum lq2*lk2) + li
//   2) qkv_gemm_kernel   : qkv_bf16[32768,2304] = X @ W_attn^T   (WMMA bf16)
//   3) diff_attn_kernel  : flash attention, double-buffered async-LDS KV
//   4) out_gemm_kernel   : out_f32[32768,768] = Y_bf16 @ W_out^T (WMMA bf16)
// ---------------------------------------------------------------------------

typedef __bf16 bf16;
typedef __attribute__((ext_vector_type(16))) __bf16 v16bf;
typedef __attribute__((ext_vector_type(8)))  float  v8f;

#define BATCH   4
#define SEQ     8192
#define NHEAD   12
#define HDIM    32
#define DVDIM   64
#define DMODEL  768
#define D3      2304

static __device__ __forceinline__ v8f wmma_bf16(v16bf a, v16bf b, v8f c) {
  // D = A(16x32 bf16) * B(32x16 bf16) + C(16x16 f32)
  return __builtin_amdgcn_wmma_f32_16x16x32_bf16(false, a, false, b,
                                                 (short)0, c, false, false);
}

static __device__ __forceinline__ float lane_reduce_max16(float v) {
#pragma unroll
  for (int m = 1; m < 16; m <<= 1) v = fmaxf(v, __shfl_xor(v, m, 32));
  return v;
}
static __device__ __forceinline__ float lane_reduce_sum16(float v) {
#pragma unroll
  for (int m = 1; m < 16; m <<= 1) v += __shfl_xor(v, m, 32);
  return v;
}

// ---- CDNA5 async global->LDS copy (ASYNCcnt path), inline asm ----
// LDS address taken from the lds_off VGPR; global address is SGPR base plus VGPR byte offset.
static __device__ __forceinline__ void async_load_b128(unsigned lds_byte_off,
                                                       unsigned gl_byte_off,
                                                       const void* base) {
  asm volatile("global_load_async_to_lds_b128 %0, %1, %2"
               :
               : "v"(lds_byte_off), "v"(gl_byte_off), "s"(base)
               : "memory");
}
static __device__ __forceinline__ void wait_asynccnt0() {
  asm volatile("s_wait_asynccnt 0" ::: "memory");
}
// Low 32 bits of a generic pointer to __shared__ == LDS byte offset
// (ISA: LDS aperture maps addr[31:0] directly to LDS).
static __device__ __forceinline__ unsigned lds_off(const void* p) {
  return (unsigned)(uintptr_t)p;
}

// ---------------------------------------------------------------------------
__global__ void lam_kernel(const float* lq1, const float* lk1,
                           const float* lq2, const float* lk2, float* lam_out) {
  if (threadIdx.x == 0) {
    float s1 = 0.f, s2 = 0.f;
    for (int i = 0; i < HDIM; ++i) { s1 += lq1[i] * lk1[i]; s2 += lq2[i] * lk2[i]; }
    const float lambda_init = 0.8f - 0.6f * __expf(-3.6f);
    lam_out[0] = __expf(s1) - __expf(s2) + lambda_init;
  }
}

// ---------------------------------------------------------------------------
// C[m, j] = sum_d X[m,d] * W[j,d], X:(32768,768) f32, W:(2304,768) f32,
// C written as bf16. Block = 256 threads = 8 waves (2x4), tile 32x64.
__global__ __launch_bounds__(256) void qkv_gemm_kernel(
    const float* __restrict__ X, const float* __restrict__ W,
    bf16* __restrict__ C) {
  const int lane = threadIdx.x & 31;
  const int wave = threadIdx.x >> 5;
  const int hi = lane >> 4;     // lane-half select
  const int ln = lane & 15;
  const int row0 = blockIdx.y * 32 + (wave >> 2) * 16;
  const int col0 = blockIdx.x * 64 + (wave & 3) * 16;
  const float* arow = X + (size_t)(row0 + ln) * DMODEL;
  const float* brow = W + (size_t)(col0 + ln) * DMODEL;
  v8f acc = {};
  for (int kk = 0; kk < DMODEL; kk += 32) {
    if (kk + 32 < DMODEL) {                       // global_prefetch_b8 next slab
      __builtin_prefetch(arow + kk + 32, 0, 1);
      __builtin_prefetch(brow + kk + 32, 0, 1);
    }
    union { v16bf v; bf16 e[16]; } A, B;
    const int abase = kk + hi * 16;  // A: k = e + hi*16
#pragma unroll
    for (int e = 0; e < 8; ++e) {
      A.e[e]     = (bf16)arow[abase + e];
      A.e[e + 8] = (bf16)arow[abase + 8 + e];
    }
    const int bbase = kk + hi * 16;  // B: k = e + hi*16
#pragma unroll
    for (int e = 0; e < 16; ++e) B.e[e] = (bf16)brow[bbase + e];
    acc = wmma_bf16(A.v, B.v, acc);
  }
#pragma unroll
  for (int i = 0; i < 8; ++i) {
    const int r = row0 + i + 8 * hi;
    C[(size_t)r * D3 + col0 + ln] = (bf16)acc[i];
  }
}

// ---------------------------------------------------------------------------
// Online-softmax update for one branch of one 32-key tile.
static __device__ __forceinline__ void softmax_update(
    v8f s0, v8f s1, float* m, float* l, v8f* o, bf16* pbuf,
    int hi, int ln, float scale) {
#pragma unroll
  for (int i = 0; i < 8; ++i) {
    const float a0 = s0[i] * scale;
    const float a1 = s1[i] * scale;
    const float rmax = lane_reduce_max16(fmaxf(a0, a1));
    const float mnew = fmaxf(m[i], rmax);
    const float alpha = __expf(m[i] - mnew);
    const float p0 = __expf(a0 - mnew);
    const float p1 = __expf(a1 - mnew);
    l[i] = l[i] * alpha + lane_reduce_sum16(p0 + p1);
    m[i] = mnew;
#pragma unroll
    for (int f = 0; f < 4; ++f) o[f][i] *= alpha;
    const int row = i + 8 * hi;
    pbuf[row * 32 + ln]      = (bf16)p0;
    pbuf[row * 32 + 16 + ln] = (bf16)p1;
  }
}

// ---------------------------------------------------------------------------
// One block = (b, h, 128 query rows). 8 waves x 16 rows.
// KV tile (32 keys: k1 32 | k2 32 | v 64 bf16) double-buffered in LDS and
// filled with global_load_async_to_lds_b128 (ASYNCcnt) one tile ahead.
__global__ __launch_bounds__(256) void diff_attn_kernel(
    const bf16* __restrict__ QKV, const float* __restrict__ lamp,
    bf16* __restrict__ Y) {
  __shared__ bf16 kv[2][32 * 128];     // [buf][key][k1 0:32 | k2 32:64 | v 64:128]
  __shared__ bf16 pb[8][2][16 * 32];   // per-wave, per-branch P staging

  const int tid  = threadIdx.x;
  const int lane = tid & 31;
  const int wave = tid >> 5;
  const int hi = lane >> 4;
  const int ln = lane & 15;

  const int bid = blockIdx.x;
  const int rb = bid & 63;                    // 64 row-blocks of 128
  const int h  = (bid >> 6) % NHEAD;
  const int b  = bid / (64 * NHEAD);

  const float lam = lamp[0];
  const float scale = 0.17677669529663687f;   // 1/sqrt(32)
  const float one_minus_li = 1.0f - (0.8f - 0.6f * __expf(-3.6f));

  const int row0 = rb * 128 + wave * 16;
  const size_t mrow = (size_t)(b * SEQ + row0 + ln) * D3;

  union Frag { v16bf v; bf16 e[16]; uint4 u[2]; };

  // Q fragments, A layout (lane=row, two contiguous 8-elem runs).
  Frag q1, q2;
  {
    const int base = hi * 8;
    const bf16* p1 = QKV + mrow + (size_t)h * 64 + base;        // branch 1
    const bf16* p2 = QKV + mrow + (size_t)h * 64 + 32 + base;   // branch 2
    q1.u[0] = *(const uint4*)(p1);
    q1.u[1] = *(const uint4*)(p1 + 16);
    q2.u[0] = *(const uint4*)(p2);
    q2.u[1] = *(const uint4*)(p2 + 16);
  }

  v8f o1[4] = {}, o2[4] = {};
  float m1[8], l1[8], m2[8], l2[8];
#pragma unroll
  for (int i = 0; i < 8; ++i) { m1[i] = -1e30f; m2[i] = -1e30f; l1[i] = 0.f; l2[i] = 0.f; }

  // Async staging coordinates: 256 threads x 32B = 8KB tile.
  const int tok   = tid >> 3;   // 32 keys, 8 threads per key
  const int chunk = tid & 7;    // 8 x 16 bf16 = 128 bf16 per key
  const unsigned joff = (chunk < 4) ? (768u + h * 64 + chunk * 16)
                                    : (1536u + h * 64 + (chunk - 4) * 16);

  auto issue_tile = [&](int t, int buf) {
    const unsigned n = (unsigned)(t * 32 + tok);
    const unsigned goff = ((unsigned)(b * SEQ) + n) * (D3 * 2u) + joff * 2u;
    const unsigned l0 = lds_off(&kv[buf][tok * 128 + chunk * 16]);
    async_load_b128(l0,      goff,      QKV);
    async_load_b128(l0 + 16, goff + 16, QKV);
  };

  issue_tile(0, 0);

  for (int t = 0; t < SEQ / 32; ++t) {
    const int cur = t & 1;
    wait_asynccnt0();        // this wave's async fills for tile t are in LDS
    __syncthreads();         // all waves' fills visible; prev reads retired

    if (t + 1 < SEQ / 32) issue_tile(t + 1, cur ^ 1);   // prefetch next tile

    const bf16* kvc = &kv[cur][0];

    // ---- S = Q K^T : two branches x two 16-key groups ----
    v8f s1[2], s2[2];
#pragma unroll
    for (int g = 0; g < 2; ++g) {
      Frag kf1, kf2;                       // B layout: lane=key col, k=(e>>3)*16+hi*8+(e&7)
      const int krow = (g * 16 + ln) * 128;
      const int hb = hi * 8;
      kf1.u[0] = *(const uint4*)(kvc + krow + hb);
      kf1.u[1] = *(const uint4*)(kvc + krow + hb + 16);
      kf2.u[0] = *(const uint4*)(kvc + krow + 32 + hb);
      kf2.u[1] = *(const uint4*)(kvc + krow + 32 + hb + 16);
      v8f z = {};
      s1[g] = wmma_bf16(q1.v, kf1.v, z);
      s2[g] = wmma_bf16(q2.v, kf2.v, z);
    }

    // ---- online softmax, both branches; P -> LDS ----
    softmax_update(s1[0], s1[1], m1, l1, o1, &pb[wave][0][0], hi, ln, scale);
    softmax_update(s2[0], s2[1], m2, l2, o2, &pb[wave][1][0], hi, ln, scale);
    __syncthreads();

    // ---- reload P in A layout ----
    Frag p1f, p2f;
    {
      const int base = hi * 8;
      const bf16* pp1 = &pb[wave][0][ln * 32 + base];
      const bf16* pp2 = &pb[wave][1][ln * 32 + base];
      p1f.u[0] = *(const uint4*)(pp1);
      p1f.u[1] = *(const uint4*)(pp1 + 16);
      p2f.u[0] = *(const uint4*)(pp2);
      p2f.u[1] = *(const uint4*)(pp2 + 16);
    }

    // ---- O += P V : V fragments shared by both branches ----
#pragma unroll
    for (int f = 0; f < 4; ++f) {
      Frag vf;                              // B layout: lane=v-col, k=key local
#pragma unroll
      for (int e = 0; e < 16; ++e) {
        const int keyl = (e & 7) + hi * 8 + (e >> 3) * 16;
        vf.e[e] = kvc[keyl * 128 + 64 + f * 16 + ln];
      }
      o1[f] = wmma_bf16(p1f.v, vf.v, o1[f]);
      o2[f] = wmma_bf16(p2f.v, vf.v, o2[f]);
    }
    // no trailing barrier: next iteration's top barrier orders reads vs refill
  }

  // ---- epilogue: y = (o1/l1 - lam * o2/l2) * (1 - lambda_init) ----
#pragma unroll
  for (int f = 0; f < 4; ++f) {
#pragma unroll
    for (int i = 0; i < 8; ++i) {
      const float v1 = o1[f][i] / l1[i];
      const float v2 = o2[f][i] / l2[i];
      const float y = (v1 - lam * v2) * one_minus_li;
      const int r = row0 + i + 8 * hi;
      Y[(size_t)(b * SEQ + r) * DMODEL + h * 64 + f * 16 + ln] = (bf16)y;
    }
  }
}

// ---------------------------------------------------------------------------
// out[m, j] = sum_d Y[m,d] * W_out[j,d], Y bf16, W f32, out f32.
__global__ __launch_bounds__(256) void out_gemm_kernel(
    const bf16* __restrict__ Yb, const float* __restrict__ W,
    float* __restrict__ Out) {
  const int lane = threadIdx.x & 31;
  const int wave = threadIdx.x >> 5;
  const int hi = lane >> 4;
  const int ln = lane & 15;
  const int row0 = blockIdx.y * 32 + (wave >> 2) * 16;
  const int col0 = blockIdx.x * 64 + (wave & 3) * 16;
  const bf16*  arow = Yb + (size_t)(row0 + ln) * DMODEL;
  const float* brow = W  + (size_t)(col0 + ln) * DMODEL;
  v8f acc = {};
  for (int kk = 0; kk < DMODEL; kk += 32) {
    if (kk + 32 < DMODEL) {
      __builtin_prefetch(arow + kk + 32, 0, 1);
      __builtin_prefetch(brow + kk + 32, 0, 1);
    }
    union { v16bf v; bf16 e[16]; uint4 u[2]; } A, B;
    const int abase = kk + hi * 16;
    A.u[0] = *(const uint4*)(arow + abase);
    A.u[1] = *(const uint4*)(arow + abase + 8);
    const int bbase = kk + hi * 16;
#pragma unroll
    for (int e = 0; e < 16; ++e) B.e[e] = (bf16)brow[bbase + e];
    acc = wmma_bf16(A.v, B.v, acc);
  }
#pragma unroll
  for (int i = 0; i < 8; ++i) {
    const int r = row0 + i + 8 * hi;
    Out[(size_t)r * DMODEL + col0 + ln] = acc[i];
  }
}

// ---------------------------------------------------------------------------
extern "C" void kernel_launch(void* const* d_in, const int* in_sizes, int n_in,
                              void* d_out, int out_size, void* d_ws, size_t ws_size,
                              hipStream_t stream) {
  const float* x      = (const float*)d_in[0];
  const float* W_attn = (const float*)d_in[1];
  const float* W_out  = (const float*)d_in[2];
  const float* lq1    = (const float*)d_in[3];
  const float* lk1    = (const float*)d_in[4];
  const float* lq2    = (const float*)d_in[5];
  const float* lk2    = (const float*)d_in[6];
  float* out = (float*)d_out;

  // Workspace layout: [lam f32 (256B slot)] [qkv bf16 151MB] [y bf16 50MB]
  char* ws = (char*)d_ws;
  float* lam = (float*)ws;
  bf16* qkv  = (bf16*)(ws + 256);
  bf16* yb   = (bf16*)(ws + 256 + (size_t)BATCH * SEQ * D3 * sizeof(bf16));

  lam_kernel<<<1, 32, 0, stream>>>(lq1, lk1, lq2, lk2, lam);

  qkv_gemm_kernel<<<dim3(D3 / 64, (BATCH * SEQ) / 32), 256, 0, stream>>>(
      x, W_attn, qkv);

  diff_attn_kernel<<<dim3(BATCH * NHEAD * (SEQ / 128)), 256, 0, stream>>>(
      qkv, lam, yb);

  out_gemm_kernel<<<dim3(DMODEL / 64, (BATCH * SEQ) / 32), 256, 0, stream>>>(
      yb, W_out, out);
}
